// TaskSpecific_25658134626781
// MI455X (gfx1250) — matrix operand order repaired and checked by reading, not compiled, not run on hardware
//
#include <hip/hip_runtime.h>

// ---------------------------------------------------------------------------
// out[b, n] = sigmoid(dot(emb[b,:], W_cat[n,:]) + b_cat[n]) * p_cat[n]
// M = 256 tasks, K = 512, N = 120581 ragged columns over 10 parameter tensors.
// f16 WMMA (16x16x32) + f32 accumulate.
// Double-buffered LDS pipeline: global loads (NT) for step it+1 in flight
// while WMMAs for step it execute; prefetch reaches it+2.
// W streamed HBM->LDS exactly once (non-temporal); emb stays L2-resident.
// ---------------------------------------------------------------------------

typedef _Float16 v16h __attribute__((ext_vector_type(16)));
typedef _Float16 h8   __attribute__((ext_vector_type(8)));
typedef float    v8f  __attribute__((ext_vector_type(8)));
typedef float    f4v  __attribute__((ext_vector_type(4)));

#define NTOT   120581
#define KDIM   512
#define MROWS  256
#define TILE_N 64
#define LDA    40   // padded f16 pitch for a 32-wide K slab (80B = 20 banks)
#define LDB    40
#define KSTEPS (KDIM / 32)

struct GateArgs {
    const float* emb;
    const float* W[10];
    const float* p[10];
    const float* b[10];
    float*       out;
};

__device__ __forceinline__ f4v ldg_nt4(const float* p) {
    return __builtin_nontemporal_load((const f4v*)p);
}

__global__ __launch_bounds__(256)
void maml_gate_wmma(GateArgs A) {
    constexpr int segStart[10] = {0, 1728, 1792, 38656, 38720,
                                  75584, 75648, 112512, 112576, 120576};

    __shared__ _Float16 Alds[2][MROWS * LDA];   // 2 x 20 KB
    __shared__ _Float16 Blds[2][TILE_N * LDB];  // 2 x  5 KB

    const int t    = threadIdx.x;
    const int wave = t >> 5;       // wave32
    const int lane = t & 31;
    const int mw   = wave >> 1;    // 0..3  (M strip of 64)
    const int nw   = wave & 1;     // 0..1  (N strip of 32)
    const int l15  = lane & 15;
    const int kh   = lane >> 4;    // K-half selector per ISA fragment layout

    const int colBase = blockIdx.x * TILE_N;

    // ---- per-thread W row pointer for B-tile staging ----
    const int j = t >> 2;          // column within tile: 0..63
    const int q = t & 3;           // k-quarter (8 floats)
    int ncol = colBase + j;
    if (ncol >= NTOT) ncol = NTOT - 1;
    const float* wrow = A.W[0];
    int woff = ncol;
#pragma unroll
    for (int s = 1; s < 10; ++s) {
        if (ncol >= segStart[s]) { wrow = A.W[s]; woff = ncol - segStart[s]; }
    }
    const float* wptr = wrow + woff * KDIM + q * 8;
    const float* aptr = A.emb + (size_t)t * KDIM;

    v8f acc[4][2];
#pragma unroll
    for (int mt = 0; mt < 4; ++mt)
#pragma unroll
        for (int nt = 0; nt < 2; ++nt)
#pragma unroll
            for (int r = 0; r < 8; ++r) acc[mt][nt][r] = 0.0f;

    // ---- staging registers ----
    f4v aS[8];    // 32 floats of emb row t
    f4v wS[2];    // 8 floats of W row (colBase+j)

    // Prologue: stage K-step 0 into buffer 0.
#pragma unroll
    for (int v = 0; v < 8; ++v) aS[v] = *(const f4v*)(aptr + 4 * v);
    wS[0] = ldg_nt4(wptr + 0);
    wS[1] = ldg_nt4(wptr + 4);
    {
        _Float16* da = &Alds[0][t * LDA];
#pragma unroll
        for (int v = 0; v < 8; ++v) {
            da[4*v+0] = (_Float16)aS[v][0];
            da[4*v+1] = (_Float16)aS[v][1];
            da[4*v+2] = (_Float16)aS[v][2];
            da[4*v+3] = (_Float16)aS[v][3];
        }
        _Float16* db = &Blds[0][j * LDB + q * 8];
#pragma unroll
        for (int e = 0; e < 4; ++e) { db[e] = (_Float16)wS[0][e]; db[e+4] = (_Float16)wS[1][e]; }
    }

    for (int it = 0; it < KSTEPS; ++it) {
        const int cur = it & 1;
        const int nxt = cur ^ 1;

        // Issue next iteration's global loads (latency hidden under WMMAs).
        if (it + 1 < KSTEPS) {
            const int k1 = (it + 1) * 32;
#pragma unroll
            for (int v = 0; v < 8; ++v) aS[v] = *(const f4v*)(aptr + k1 + 4 * v);
            wS[0] = ldg_nt4(wptr + k1 + 0);
            wS[1] = ldg_nt4(wptr + k1 + 4);
        }
        if (it + 2 < KSTEPS) {
            __builtin_prefetch(wptr + (it + 2) * 32, 0, 0);   // global_prefetch_b8
        }

        __syncthreads();   // buf[cur] stores visible; prior readers of buf[nxt] done

        // ---- fragments per ISA 16-bit layouts, from buf[cur] ----
        const _Float16* Ab = Alds[cur];
        const _Float16* Bb = Blds[cur];
        v16h af[4], bf[2];
#pragma unroll
        for (int mt = 0; mt < 4; ++mt) {
            const int mr = mw * 64 + mt * 16 + l15;
            h8 lo = *(const h8*)&Ab[mr * LDA + kh * 8];        // K = 8*kh .. +7
            h8 hi = *(const h8*)&Ab[mr * LDA + 16 + kh * 8];   // K = 16+8*kh .. +7
#pragma unroll
            for (int e = 0; e < 8; ++e) { af[mt][e] = lo[e]; af[mt][e + 8] = hi[e]; }
        }
#pragma unroll
        for (int nt = 0; nt < 2; ++nt) {
            const int nc = nw * 32 + nt * 16 + l15;
            h8 lo = *(const h8*)&Bb[nc * LDB + kh * 8];        // K = 8*kh .. +7
            h8 hi = *(const h8*)&Bb[nc * LDB + 16 + kh * 8];   // K = 16+8*kh .. +7
#pragma unroll
            for (int e = 0; e < 8; ++e) { bf[nt][e] = lo[e]; bf[nt][e + 8] = hi[e]; }
        }

        // ---- 8 WMMAs: 256x64 tile advances 32 in K ----
#pragma unroll
        for (int mt = 0; mt < 4; ++mt)
#pragma unroll
            for (int nt = 0; nt < 2; ++nt)
                acc[mt][nt] = __builtin_amdgcn_wmma_f32_16x16x32_f16(
                    false, af[mt], false, bf[nt],
                    (short)0, acc[mt][nt], false, false);

        // ---- convert + store staged data into buf[nxt] ----
        if (it + 1 < KSTEPS) {
            _Float16* da = &Alds[nxt][t * LDA];
#pragma unroll
            for (int v = 0; v < 8; ++v) {
                da[4*v+0] = (_Float16)aS[v][0];
                da[4*v+1] = (_Float16)aS[v][1];
                da[4*v+2] = (_Float16)aS[v][2];
                da[4*v+3] = (_Float16)aS[v][3];
            }
            _Float16* db = &Blds[nxt][j * LDB + q * 8];
#pragma unroll
            for (int e = 0; e < 4; ++e) { db[e] = (_Float16)wS[0][e]; db[e+4] = (_Float16)wS[1][e]; }
        }
    }

    // ---- epilogue: sigmoid(acc + bias) * p, ragged segment lookup ----
#pragma unroll
    for (int nt = 0; nt < 2; ++nt) {
        const int  n     = colBase + nw * 32 + nt * 16 + l15;
        const bool valid = (n < NTOT);
        const int  ncl   = valid ? n : (NTOT - 1);
        const float* pp = A.p[0];
        const float* bb = A.b[0];
        int off = ncl;
#pragma unroll
        for (int s = 1; s < 10; ++s) {
            if (ncl >= segStart[s]) { pp = A.p[s]; bb = A.b[s]; off = ncl - segStart[s]; }
        }
        const float pv = pp[off];
        const float bv = bb[off];
#pragma unroll
        for (int mt = 0; mt < 4; ++mt) {
#pragma unroll
            for (int r = 0; r < 8; ++r) {
                const int m  = mw * 64 + mt * 16 + kh * 8 + r;  // C/D VGPR r layout
                float x  = acc[mt][nt][r] + bv;
                float sg = 1.0f / (1.0f + __expf(-x));
                if (valid) {
                    __builtin_nontemporal_store(sg * pv, &A.out[(size_t)m * NTOT + n]);
                }
            }
        }
    }
}

extern "C" void kernel_launch(void* const* d_in, const int* in_sizes, int n_in,
                              void* d_out, int out_size, void* d_ws, size_t ws_size,
                              hipStream_t stream) {
    (void)in_sizes; (void)n_in; (void)out_size; (void)d_ws; (void)ws_size;
    GateArgs A;
    A.emb = (const float*)d_in[0];
    for (int i = 0; i < 10; ++i) {
        A.p[i] = (const float*)d_in[1 + 3 * i];
        A.W[i] = (const float*)d_in[2 + 3 * i];
        A.b[i] = (const float*)d_in[3 + 3 * i];
    }
    A.out = (float*)d_out;

    const int nTiles = (NTOT + TILE_N - 1) / TILE_N;   // 1885 blocks
    maml_gate_wmma<<<nTiles, 256, 0, stream>>>(A);
}
